// CausalSelfAttention_47914655154685
// MI455X (gfx1250) — matrix operand order repaired and checked by reading, not compiled, not run on hardware
//
#include <hip/hip_runtime.h>
#include <hip/hip_bf16.h>

typedef unsigned short u16;
typedef __attribute__((ext_vector_type(16))) __bf16 v16bf;
typedef __attribute__((ext_vector_type(8)))  float  v8f;
typedef __attribute__((ext_vector_type(4)))  unsigned int v4u;

#define B_   4
#define T_   2048
#define C_   1024
#define H_   16
#define D_   64
#define C3_  3072

__device__ __forceinline__ u16 f2bf(float f) {
    unsigned int u = __float_as_uint(f);
    u += 0x7FFFu + ((u >> 16) & 1u);
    return (u16)(u >> 16);
}

__device__ __forceinline__ v16bf load_frag(const u16* p0, const u16* p1) {
    v16bf a;
    ((v4u*)&a)[0] = *(const v4u*)p0;
    ((v4u*)&a)[1] = *(const v4u*)p1;
    return a;
}

__device__ __forceinline__ v8f vzero8() {
    v8f z = {0.f, 0.f, 0.f, 0.f, 0.f, 0.f, 0.f, 0.f};
    return z;
}

// ---------------- fp32 -> bf16 convert ----------------
__global__ __launch_bounds__(256) void k_f32_to_bf16(const float* __restrict__ in,
                                                     u16* __restrict__ out, int n) {
    int i = blockIdx.x * 256 + threadIdx.x;
    if (i < n) out[i] = f2bf(in[i]);
}

// ---------------- transpose + convert: W[K][N] -> Wt[N][K] bf16 ----------------
__global__ __launch_bounds__(256) void k_transpose_bf16(const float* __restrict__ W,
                                                        u16* __restrict__ Wt,
                                                        int K, int N) {
    int k = blockIdx.x * 256 + threadIdx.x;  // coalesced writes along k
    int n = blockIdx.y;
    if (k < K) Wt[(size_t)n * K + k] = f2bf(W[(size_t)k * N + n]);
}

// ---------------- bf16 WMMA GEMM: C[M,N] = A[M,K] * Bt[N,K]^T ----------------
// block tile 128x128, 8 waves as 4(m) x 2(n), wave tile 32x64 (2x4 wmma accums)
__global__ __launch_bounds__(256) void k_gemm_bf16(const u16* __restrict__ A,
                                                   const u16* __restrict__ Bt,
                                                   float* __restrict__ Cmat,
                                                   int M, int N, int K) {
    const int wave = threadIdx.x >> 5;
    const int lane = threadIdx.x & 31;
    const int hi   = lane >> 4;
    const int l16  = lane & 15;
    const int wm = wave >> 1, wn = wave & 1;
    const int m0 = blockIdx.x * 128 + wm * 32;
    const int n0 = blockIdx.y * 128 + wn * 64;

    v8f acc[2][4];
#pragma unroll
    for (int mi = 0; mi < 2; mi++)
#pragma unroll
        for (int ni = 0; ni < 4; ni++)
            acc[mi][ni] = vzero8();

    const u16* ap[2];
    const u16* bp[4];
#pragma unroll
    for (int mi = 0; mi < 2; mi++)
        ap[mi] = A + (size_t)(m0 + mi * 16 + l16) * K + 8 * hi;
#pragma unroll
    for (int ni = 0; ni < 4; ni++)
        bp[ni] = Bt + (size_t)(n0 + ni * 16 + l16) * K + 8 * hi;

    for (int kb = 0; kb < K; kb += 32) {
        // pull next lines into all cache levels (WGP scope) -- waves share rows
        __builtin_prefetch(ap[0] + kb + 256, 0, 3);
        __builtin_prefetch(bp[0] + kb + 256, 0, 3);
        __builtin_prefetch(bp[2] + kb + 256, 0, 3);
        v16bf a[2], b[4];
#pragma unroll
        for (int mi = 0; mi < 2; mi++)
            a[mi] = load_frag(ap[mi] + kb, ap[mi] + kb + 16);
#pragma unroll
        for (int ni = 0; ni < 4; ni++)
            b[ni] = load_frag(bp[ni] + kb, bp[ni] + kb + 16);
#pragma unroll
        for (int mi = 0; mi < 2; mi++)
#pragma unroll
            for (int ni = 0; ni < 4; ni++)
                acc[mi][ni] = __builtin_amdgcn_wmma_f32_16x16x32_bf16(
                    false, a[mi], false, b[ni], (short)0, acc[mi][ni], false, false);
    }

#pragma unroll
    for (int mi = 0; mi < 2; mi++)
#pragma unroll
        for (int ni = 0; ni < 4; ni++)
#pragma unroll
            for (int r = 0; r < 8; r++) {
                int m = m0 + mi * 16 + hi * 8 + r;
                int n = n0 + ni * 16 + l16;
                Cmat[(size_t)m * N + n] = acc[mi][ni][r];
            }
}

// ---------------- RMSNorm + RoPE + repack q,k,v to bf16 layouts ----------------
// thread per (bh, t), t fastest. q,k -> [BH][T][D] bf16; v -> [BH][D][T] bf16.
__global__ __launch_bounds__(256) void k_qkv_post(const float* __restrict__ qkv,
                                                  const float* __restrict__ qw,
                                                  const float* __restrict__ kw,
                                                  u16* __restrict__ qb,
                                                  u16* __restrict__ kb,
                                                  u16* __restrict__ vt) {
    int idx = blockIdx.x * 256 + threadIdx.x;
    int t  = idx & (T_ - 1);
    int bh = idx >> 11;
    int b  = bh >> 4, h = bh & 15;
    const float* row = qkv + ((size_t)(b * T_ + t)) * C3_ + h * D_;

    float q[64], k[64];
#pragma unroll
    for (int i = 0; i < 16; i++) {
        float4 fq = *(const float4*)(row + i * 4);
        float4 fk = *(const float4*)(row + C_ + i * 4);
        q[i*4+0]=fq.x; q[i*4+1]=fq.y; q[i*4+2]=fq.z; q[i*4+3]=fq.w;
        k[i*4+0]=fk.x; k[i*4+1]=fk.y; k[i*4+2]=fk.z; k[i*4+3]=fk.w;
    }
    float sq = 0.f, sk = 0.f;
#pragma unroll
    for (int d = 0; d < 64; d++) { sq += q[d]*q[d]; sk += k[d]*k[d]; }
    float rq = rsqrtf(sq * (1.f/64.f) + 1e-5f);
    float rk = rsqrtf(sk * (1.f/64.f) + 1e-5f);

    alignas(16) u16 qo[64];
    alignas(16) u16 ko[64];
#pragma unroll
    for (int d = 0; d < 32; d++) {
        // inv_freq = 10000^(-2d/64) = exp(-2d/64 * ln(10000))
        float inv = __expf((float)(-2 * d) * (9.210340371976184f / 64.f));
        float s, c;
        __sincosf((float)t * inv, &s, &c);
        float q1 = q[d]      * rq * qw[d];
        float q2 = q[d + 32] * rq * qw[d + 32];
        float k1 = k[d]      * rk * kw[d];
        float k2 = k[d + 32] * rk * kw[d + 32];
        // fold softmax scale 1/sqrt(64) into q
        qo[d]      = f2bf((q1 * c - q2 * s) * 0.125f);
        qo[d + 32] = f2bf((q2 * c + q1 * s) * 0.125f);
        ko[d]      = f2bf(k1 * c - k2 * s);
        ko[d + 32] = f2bf(k2 * c + k1 * s);
    }
    u16* qrow = qb + ((size_t)bh * T_ + t) * D_;
    u16* krow = kb + ((size_t)bh * T_ + t) * D_;
#pragma unroll
    for (int i = 0; i < 8; i++) {
        ((v4u*)qrow)[i] = ((const v4u*)qo)[i];
        ((v4u*)krow)[i] = ((const v4u*)ko)[i];
    }
    const float* vrow = row + 2 * C_;
    u16* vcol = vt + (size_t)bh * D_ * T_ + t;
#pragma unroll
    for (int d = 0; d < 64; d++) vcol[(size_t)d * T_] = f2bf(vrow[d]);
}

// ---------------- causal flash attention, bf16 WMMA, online softmax ----------------
// grid: (T/128, B*H); 8 waves/block, each wave owns 16 q rows.
__global__ __launch_bounds__(256) void k_flash(const u16* __restrict__ qb,
                                               const u16* __restrict__ kb,
                                               const u16* __restrict__ vtb,
                                               u16* __restrict__ yb) {
    __shared__ u16 Plds[8][16 * 64];
    const int wave = threadIdx.x >> 5;
    const int lane = threadIdx.x & 31;
    const int hi   = lane >> 4;
    const int l16  = lane & 15;
    const int bh = blockIdx.y;
    const int q0 = blockIdx.x * 128 + wave * 16;

    const u16* Q  = qb  + (size_t)bh * T_ * D_;
    const u16* K  = kb  + (size_t)bh * T_ * D_;
    const u16* Vt = vtb + (size_t)bh * D_ * T_;

    // Q A-fragments for both 32-wide d chunks (held for the whole kernel)
    v16bf qf[2];
#pragma unroll
    for (int kc = 0; kc < 2; kc++) {
        const u16* qp = Q + (size_t)(q0 + l16) * D_ + kc * 32 + 8 * hi;
        qf[kc] = load_frag(qp, qp + 16);
    }

    v8f o[4];
    o[0] = vzero8(); o[1] = vzero8(); o[2] = vzero8(); o[3] = vzero8();
    float mrow[8], lrow[8];
#pragma unroll
    for (int r = 0; r < 8; r++) { mrow[r] = -1e30f; lrow[r] = 0.f; }

    const int kend = q0 + 16;  // causal bound for this wave's rows
    for (int kt = 0; kt < kend; kt += 64) {
        // ---- S = Q K^T over this 64-wide k tile ----
        v8f s[4];
        s[0] = vzero8(); s[1] = vzero8(); s[2] = vzero8(); s[3] = vzero8();
#pragma unroll
        for (int j = 0; j < 4; j++) {
#pragma unroll
            for (int kc = 0; kc < 2; kc++) {
                const u16* kp = K + (size_t)(kt + j * 16 + l16) * D_ + kc * 32 + 8 * hi;
                v16bf kf = load_frag(kp, kp + 16);
                s[j] = __builtin_amdgcn_wmma_f32_16x16x32_bf16(false, qf[kc], false, kf, (short)0, s[j], false, false);
            }
        }
        // ---- causal mask ----
#pragma unroll
        for (int j = 0; j < 4; j++) {
            int kg = kt + j * 16 + l16;
#pragma unroll
            for (int r = 0; r < 8; r++) {
                int qg = q0 + hi * 8 + r;
                if (kg > qg) s[j][r] = -1e30f;
            }
        }
        // ---- online softmax row stats (rows live across 16-lane halves) ----
#pragma unroll
        for (int r = 0; r < 8; r++) {
            float mx = s[0][r];
            mx = fmaxf(mx, s[1][r]); mx = fmaxf(mx, s[2][r]); mx = fmaxf(mx, s[3][r]);
            mx = fmaxf(mx, __shfl_xor(mx, 1));
            mx = fmaxf(mx, __shfl_xor(mx, 2));
            mx = fmaxf(mx, __shfl_xor(mx, 4));
            mx = fmaxf(mx, __shfl_xor(mx, 8));
            float mnew = fmaxf(mrow[r], mx);
            float corr = __expf(mrow[r] - mnew);
            mrow[r] = mnew;
            lrow[r] *= corr;
#pragma unroll
            for (int nt = 0; nt < 4; nt++) o[nt][r] *= corr;
            float rs = 0.f;
#pragma unroll
            for (int j = 0; j < 4; j++) {
                float p = __expf(s[j][r] - mnew);
                s[j][r] = p;
                rs += p;
            }
            rs += __shfl_xor(rs, 1);
            rs += __shfl_xor(rs, 2);
            rs += __shfl_xor(rs, 4);
            rs += __shfl_xor(rs, 8);
            lrow[r] += rs;
        }
        // ---- P: C-layout -> A-layout via wave-private LDS patch ----
        u16* pl = &Plds[wave][0];
#pragma unroll
        for (int j = 0; j < 4; j++)
#pragma unroll
            for (int r = 0; r < 8; r++)
                pl[(hi * 8 + r) * 64 + j * 16 + l16] = f2bf(s[j][r]);
        // (DS ops are in-order within a wave; compiler inserts the load waits)
        // ---- O += P @ V (V pre-transposed [D][T] -> contiguous B frags) ----
#pragma unroll
        for (int kc = 0; kc < 2; kc++) {
            const u16* pp = pl + (size_t)l16 * 64 + kc * 32 + 8 * hi;
            v16bf pf = load_frag(pp, pp + 16);
#pragma unroll
            for (int nt = 0; nt < 4; nt++) {
                const u16* vp = Vt + (size_t)(nt * 16 + l16) * T_ + kt + kc * 32 + 8 * hi;
                v16bf vf = load_frag(vp, vp + 16);
                o[nt] = __builtin_amdgcn_wmma_f32_16x16x32_bf16(false, pf, false, vf, (short)0, o[nt], false, false);
            }
        }
    }
    // ---- normalize and write y as bf16 in [B,T,H*D] for the projection GEMM ----
    int b = bh >> 4, h = bh & 15;
#pragma unroll
    for (int nt = 0; nt < 4; nt++)
#pragma unroll
        for (int r = 0; r < 8; r++) {
            int tq = q0 + hi * 8 + r;
            int d  = nt * 16 + l16;
            yb[((size_t)(b * T_ + tq)) * C_ + h * D_ + d] = f2bf(o[nt][r] / lrow[r]);
        }
}

extern "C" void kernel_launch(void* const* d_in, const int* in_sizes, int n_in,
                              void* d_out, int out_size, void* d_ws, size_t ws_size,
                              hipStream_t stream) {
    (void)in_sizes; (void)n_in; (void)out_size; (void)ws_size;
    const float* x     = (const float*)d_in[0];
    const float* Wqkv  = (const float*)d_in[1];
    const float* Wproj = (const float*)d_in[2];
    const float* qw    = (const float*)d_in[3];
    const float* kw    = (const float*)d_in[4];
    float* out = (float*)d_out;

    char* ws = (char*)d_ws;
    size_t off = 0;
    auto alloc = [&](size_t bytes) -> void* {
        void* p = ws + off;
        off = (off + bytes + 255) & ~(size_t)255;
        return p;
    };
    const size_t BT = (size_t)B_ * T_;              // 8192
    const size_t BHTD = (size_t)B_ * H_ * T_ * D_;  // 8388608

    u16*   xb     = (u16*)alloc(BT * C_ * 2);
    u16*   wqkvT  = (u16*)alloc((size_t)C3_ * C_ * 2);
    u16*   wprojT = (u16*)alloc((size_t)C_ * C_ * 2);
    float* qkv    = (float*)alloc(BT * C3_ * 4);
    u16*   qbuf   = (u16*)alloc(BHTD * 2);
    u16*   kbuf   = (u16*)alloc(BHTD * 2);
    u16*   vtbuf  = (u16*)alloc(BHTD * 2);
    u16*   ybuf   = (u16*)alloc(BT * C_ * 2);

    // 1) convert x to bf16; transpose+convert weights
    k_f32_to_bf16<<<(int)((BT * C_) / 256), 256, 0, stream>>>(x, xb, (int)(BT * C_));
    k_transpose_bf16<<<dim3(C_ / 256, C3_), 256, 0, stream>>>(Wqkv, wqkvT, C_, C3_);
    k_transpose_bf16<<<dim3(C_ / 256, C_),  256, 0, stream>>>(Wproj, wprojT, C_, C_);
    // 2) qkv = x @ Wqkv   [8192 x 3072], block tile 128x128
    k_gemm_bf16<<<dim3(64, 24), 256, 0, stream>>>(xb, wqkvT, qkv, (int)BT, C3_, C_);
    // 3) RMSNorm + RoPE + repack
    k_qkv_post<<<(int)((size_t)B_ * H_ * T_ / 256), 256, 0, stream>>>(qkv, qw, kw, qbuf, kbuf, vtbuf);
    // 4) causal flash attention
    k_flash<<<dim3(T_ / 128, B_ * H_), 256, 0, stream>>>(qbuf, kbuf, vtbuf, ybuf);
    // 5) out = y @ Wproj  [8192 x 1024], block tile 128x128
    k_gemm_bf16<<<dim3(64, 8), 256, 0, stream>>>(ybuf, wprojT, out, (int)BT, C_, C_);
}
